// Extractor_bimodal_32272384262573
// MI455X (gfx1250) — matrix operand order repaired and checked by reading, not compiled, not run on hardware
//
#include <hip/hip_runtime.h>
#include <hip/hip_bf16.h>

typedef __attribute__((ext_vector_type(16))) _Float16 v16h;
typedef __attribute__((ext_vector_type(8)))  float    v8f;

#define DIMC 768
#define NHC 6
#define NPC 4
#define DHC 128
#define HIDC 192
#define NQC 21504
#define NFC 4096
#define KB768 24
#define KB192 6

// ---------------------------------------------------------------------------
// WMMA wave32 fragment index helpers (CDNA5 ISA 7.12.2 layouts)
// A 16x32 f16 fragment: lane = h*16 + m%16 ; element e:
//   e<8  -> K = e + 8*h        (K 0..15)
//   e>=8 -> K = 16+(e-8)+8*h   (K 16..31)
// B 32x16 f16 fragment: lane = ((K/8)%2)*16 + n%16 ; element e = K%8 + 8*(K/16)
// ---------------------------------------------------------------------------
__device__ __forceinline__ size_t apack_idx(int m, int k, int KB) {
    int mt = m >> 4, mr = m & 15;
    int kb = k >> 5, ko = k & 31;
    int h, e;
    if (ko < 16) { h = ko >> 3;      e = ko & 7; }
    else         { int t = ko - 16; h = t >> 3; e = 8 + (t & 7); }
    return (((size_t)mt * KB + kb) * 32 + (h * 16 + mr)) * 16 + e;
}

__device__ __forceinline__ size_t bpack_idx(int k, int n, int KB) {
    int nt = n >> 4, nr = n & 15;
    int kb = k >> 5, ko = k & 31;
    int lg = (ko >> 3) & 1, e = (ko & 7) | ((ko >> 4) << 3);
    return (((size_t)nt * KB + kb) * 32 + (lg * 16 + nr)) * 16 + e;
}

// ---------------------------------------------------------------------------
// Weight f32 (K x N row-major) -> packed f16 B-fragments, N padded with zeros
// ---------------------------------------------------------------------------
__global__ void pack_w_kernel(const float* __restrict__ W, _Float16* __restrict__ Bp,
                              int K, int N, int Npad) {
    int idx = blockIdx.x * blockDim.x + threadIdx.x;
    if (idx >= K * Npad) return;
    int k = idx / Npad, n = idx % Npad;
    float v = (n < N) ? W[(size_t)k * N + n] : 0.0f;
    Bp[bpack_idx(k, n, K >> 5)] = (_Float16)v;
}

// ---------------------------------------------------------------------------
// LayerNorm over K=768 + pack into A-fragment layout. One block (256t) per row.
// ---------------------------------------------------------------------------
__global__ void ln_pack_kernel(const float* __restrict__ X, const float* __restrict__ g,
                               const float* __restrict__ b, _Float16* __restrict__ Ap) {
    __shared__ float red[256];
    int m = blockIdx.x, tid = threadIdx.x;
    const float* row = X + (size_t)m * DIMC;
    float x0 = row[tid], x1 = row[tid + 256], x2 = row[tid + 512];
    red[tid] = x0 + x1 + x2;
    __syncthreads();
    for (int off = 128; off > 0; off >>= 1) {
        if (tid < off) red[tid] += red[tid + off];
        __syncthreads();
    }
    float mean = red[0] * (1.0f / DIMC);
    __syncthreads();
    float d0 = x0 - mean, d1 = x1 - mean, d2 = x2 - mean;
    red[tid] = d0 * d0 + d1 * d1 + d2 * d2;
    __syncthreads();
    for (int off = 128; off > 0; off >>= 1) {
        if (tid < off) red[tid] += red[tid + off];
        __syncthreads();
    }
    float rstd = rsqrtf(red[0] * (1.0f / DIMC) + 1e-6f);
    float d[3] = {d0, d1, d2};
    for (int i = 0; i < 3; ++i) {
        int k = tid + i * 256;
        float y = d[i] * rstd * g[k] + b[k];
        Ap[apack_idx(m, k, KB768)] = (_Float16)y;
    }
}

// ---------------------------------------------------------------------------
// WMMA GEMM: C[M x N] = Apacked * Bpacked + bias (+ residual)
// grid = (M/128, Npad/64); block = dim3(32,4): 4 waves, each wave owns a
// 32x64 strip of C (2 M-tiles x 4 N-tiles = 8 accumulators).
// Software-pipelined: fragments for step kb+1 are fetched before the 8 WMMAs
// of step kb so the load clause overlaps the WMMA dependency chain.
// ---------------------------------------------------------------------------
#define WMMA_F16(A, B, Cc) \
    __builtin_amdgcn_wmma_f32_16x16x32_f16(false, (A), false, (B), (short)0, (Cc), false, false)

__global__ void __launch_bounds__(128) gemm_wmma_kernel(
        const _Float16* __restrict__ Ap, const _Float16* __restrict__ Bp,
        const float* __restrict__ bias, const float* __restrict__ res,
        float* __restrict__ C, int N, int KB) {
    const int lane = threadIdx.x;
    const int mt2  = (blockIdx.x * 4 + threadIdx.y) * 2;  // first of two 16-row tiles
    const int ntb  = blockIdx.y;

    const _Float16* apA = Ap + ((size_t)(mt2 + 0) * KB * 32 + lane) * 16;
    const _Float16* apB = Ap + ((size_t)(mt2 + 1) * KB * 32 + lane) * 16;
    const _Float16* bp0 = Bp + ((size_t)(ntb * 4 + 0) * KB * 32 + lane) * 16;
    const _Float16* bp1 = Bp + ((size_t)(ntb * 4 + 1) * KB * 32 + lane) * 16;
    const _Float16* bp2 = Bp + ((size_t)(ntb * 4 + 2) * KB * 32 + lane) * 16;
    const _Float16* bp3 = Bp + ((size_t)(ntb * 4 + 3) * KB * 32 + lane) * 16;

    v8f acc[8];
    #pragma unroll
    for (int i = 0; i < 8; ++i) acc[i] = v8f{0,0,0,0,0,0,0,0};

    // prologue: fragments for kb = 0
    v16h a0 = *(const v16h*)apA;
    v16h a1 = *(const v16h*)apB;
    v16h b0 = *(const v16h*)bp0;
    v16h b1 = *(const v16h*)bp1;
    v16h b2 = *(const v16h*)bp2;
    v16h b3 = *(const v16h*)bp3;

    for (int kb = 0; kb < KB; ++kb) {
        const int kn = (kb + 1 < KB) ? kb + 1 : kb;   // last iter: benign reload
        const size_t o = (size_t)kn * 512;            // 32 lanes * 16 halfs per k-step
        v16h na0 = *(const v16h*)(apA + o);
        v16h na1 = *(const v16h*)(apB + o);
        v16h nb0 = *(const v16h*)(bp0 + o);
        v16h nb1 = *(const v16h*)(bp1 + o);
        v16h nb2 = *(const v16h*)(bp2 + o);
        v16h nb3 = *(const v16h*)(bp3 + o);

        acc[0] = WMMA_F16(a0, b0, acc[0]);
        acc[1] = WMMA_F16(a1, b0, acc[1]);
        acc[2] = WMMA_F16(a0, b1, acc[2]);
        acc[3] = WMMA_F16(a1, b1, acc[3]);
        acc[4] = WMMA_F16(a0, b2, acc[4]);
        acc[5] = WMMA_F16(a1, b2, acc[5]);
        acc[6] = WMMA_F16(a0, b3, acc[6]);
        acc[7] = WMMA_F16(a1, b3, acc[7]);

        a0 = na0; a1 = na1;
        b0 = nb0; b1 = nb1; b2 = nb2; b3 = nb3;
    }

    const int nlane = lane & 15;
    const int half8 = (lane >> 4) << 3;
    #pragma unroll
    for (int t = 0; t < 4; ++t) {
        int n = ntb * 64 + t * 16 + nlane;
        if (n >= N) continue;
        float bv = bias[n];
        #pragma unroll
        for (int mi = 0; mi < 2; ++mi) {
            int mbase = (mt2 + mi) * 16 + half8;
            v8f av = acc[t * 2 + mi];
            #pragma unroll
            for (int r = 0; r < 8; ++r) {
                size_t idx = (size_t)(mbase + r) * N + n;
                float val = av[r] + bv;
                if (res) val += res[idx];
                C[idx] = val;
            }
        }
    }
}

// ---------------------------------------------------------------------------
// Deformable sampling: softmax over 4 points + bilinear gather from v,
// writes result directly in packed A-fragment f16 layout (K = 768).
// One wave per (query, head); lane handles 4 consecutive channels.
// ---------------------------------------------------------------------------
__global__ void msdeform_sample_kernel(const float* __restrict__ v,
                                       const float* __restrict__ off,
                                       const float* __restrict__ awl,
                                       const float* __restrict__ refp,
                                       _Float16* __restrict__ Ap) {
    int tid  = threadIdx.x;
    int wid  = blockIdx.x * 8 + (tid >> 5);
    int lane = tid & 31;
    int q = wid / NHC, h = wid % NHC;
    if (q >= NQC) return;

    float rx = refp[q * 2 + 0], ry = refp[q * 2 + 1];

    float l0 = awl[q * 24 + h * 4 + 0];
    float l1 = awl[q * 24 + h * 4 + 1];
    float l2 = awl[q * 24 + h * 4 + 2];
    float l3 = awl[q * 24 + h * 4 + 3];
    float mx = fmaxf(fmaxf(l0, l1), fmaxf(l2, l3));
    float e0 = __expf(l0 - mx), e1 = __expf(l1 - mx);
    float e2 = __expf(l2 - mx), e3 = __expf(l3 - mx);
    float inv = 1.0f / (e0 + e1 + e2 + e3);
    float sw[4] = {e0 * inv, e1 * inv, e2 * inv, e3 * inv};

    float a0 = 0.f, a1 = 0.f, a2 = 0.f, a3 = 0.f;
    const size_t coff = (size_t)h * DHC + (size_t)lane * 4;

    for (int p = 0; p < NPC; ++p) {
        float ox = off[q * 48 + h * 8 + p * 2 + 0];
        float oy = off[q * 48 + h * 8 + p * 2 + 1];
        float px = rx * 64.0f + ox - 0.5f;
        float py = ry * 64.0f + oy - 0.5f;
        float x0f = floorf(px), y0f = floorf(py);
        float fx = px - x0f, fy = py - y0f;
        int ix0 = (int)x0f, iy0 = (int)y0f;
        float s = sw[p];

        #pragma unroll
        for (int c = 0; c < 4; ++c) {
            int ix = ix0 + (c & 1);
            int iy = iy0 + (c >> 1);
            float wt = ((c & 1) ? fx : 1.0f - fx) * ((c >> 1) ? fy : 1.0f - fy);
            bool valid = (ix >= 0) & (ix < 64) & (iy >= 0) & (iy < 64);
            int cx = ix < 0 ? 0 : (ix > 63 ? 63 : ix);
            int cy = iy < 0 ? 0 : (iy > 63 ? 63 : iy);
            const float4 val = *(const float4*)(v + (size_t)(cy * 64 + cx) * DIMC + coff);
            float ww = valid ? s * wt : 0.0f;
            a0 += ww * val.x; a1 += ww * val.y; a2 += ww * val.z; a3 += ww * val.w;
        }
    }

    int kbase = h * DHC + lane * 4;
    Ap[apack_idx(q, kbase + 0, KB768)] = (_Float16)a0;
    Ap[apack_idx(q, kbase + 1, KB768)] = (_Float16)a1;
    Ap[apack_idx(q, kbase + 2, KB768)] = (_Float16)a2;
    Ap[apack_idx(q, kbase + 3, KB768)] = (_Float16)a3;
}

// ---------------------------------------------------------------------------
// Depthwise 3x3 conv over 3 spatial segments + bias + exact GELU,
// writes packed A-fragment f16 layout (K = 192). Thread = (token, 4 channels).
// ---------------------------------------------------------------------------
__global__ void dwconv_gelu_pack_kernel(const float* __restrict__ X,
                                        const float* __restrict__ k9,
                                        const float* __restrict__ bias,
                                        _Float16* __restrict__ Ap) {
    int gid = blockIdx.x * blockDim.x + threadIdx.x;
    if (gid >= NQC * 48) return;
    int t = gid / 48, c0 = (gid % 48) * 4;

    int hh, ww, base;
    if (t < 16384)      { hh = 128; ww = 128; base = 0; }
    else if (t < 20480) { hh = 64;  ww = 64;  base = 16384; }
    else                { hh = 32;  ww = 32;  base = 20480; }
    int loc = t - base;
    int i = loc / ww, j = loc % ww;

    float acc[4] = {bias[c0], bias[c0 + 1], bias[c0 + 2], bias[c0 + 3]};
    for (int dy = 0; dy < 3; ++dy) {
        int ii = i + dy - 1;
        if (ii < 0 || ii >= hh) continue;
        for (int dx = 0; dx < 3; ++dx) {
            int jj = j + dx - 1;
            if (jj < 0 || jj >= ww) continue;
            const float* xp = X + (size_t)(base + ii * ww + jj) * HIDC + c0;
            #pragma unroll
            for (int u = 0; u < 4; ++u)
                acc[u] += k9[(c0 + u) * 9 + dy * 3 + dx] * xp[u];
        }
    }
    #pragma unroll
    for (int u = 0; u < 4; ++u) {
        float x = acc[u];
        float gx = 0.5f * x * (1.0f + erff(x * 0.70710678118654752f));
        Ap[apack_idx(t, c0 + u, KB192)] = (_Float16)gx;
    }
}

// ---------------------------------------------------------------------------
// Launch
// ---------------------------------------------------------------------------
static inline char* ws_carve(char*& p, size_t bytes) {
    char* r = p;
    p += (bytes + 255) & ~(size_t)255;
    return r;
}

extern "C" void kernel_launch(void* const* d_in, const int* in_sizes, int n_in,
                              void* d_out, int out_size, void* d_ws, size_t ws_size,
                              hipStream_t stream) {
    const float* query = (const float*)d_in[0];
    const float* qom   = (const float*)d_in[1];
    const float* refp  = (const float*)d_in[2];
    const float* feat  = (const float*)d_in[3];
    const float* fn_g  = (const float*)d_in[4];
    const float* fn_b  = (const float*)d_in[5];
    const float* qn_g[2]  = {(const float*)d_in[6],  (const float*)d_in[8]};
    const float* qn_b[2]  = {(const float*)d_in[7],  (const float*)d_in[9]};
    const float* val_w[2] = {(const float*)d_in[10], (const float*)d_in[18]};
    const float* val_b[2] = {(const float*)d_in[11], (const float*)d_in[19]};
    const float* off_w[2] = {(const float*)d_in[12], (const float*)d_in[20]};
    const float* off_b[2] = {(const float*)d_in[13], (const float*)d_in[21]};
    const float* aw_w[2]  = {(const float*)d_in[14], (const float*)d_in[22]};
    const float* aw_b[2]  = {(const float*)d_in[15], (const float*)d_in[23]};
    const float* out_w[2] = {(const float*)d_in[16], (const float*)d_in[24]};
    const float* out_b[2] = {(const float*)d_in[17], (const float*)d_in[25]};
    const float* fc1_w[2] = {(const float*)d_in[26], (const float*)d_in[32]};
    const float* fc1_b[2] = {(const float*)d_in[27], (const float*)d_in[33]};
    const float* dw_k[2]  = {(const float*)d_in[28], (const float*)d_in[34]};
    const float* dw_b[2]  = {(const float*)d_in[29], (const float*)d_in[35]};
    const float* fc2_w[2] = {(const float*)d_in[30], (const float*)d_in[36]};
    const float* fc2_b[2] = {(const float*)d_in[31], (const float*)d_in[37]};
    const float* ffn_g[2] = {(const float*)d_in[38], (const float*)d_in[40]};
    const float* ffn_b[2] = {(const float*)d_in[39], (const float*)d_in[41]};
    const float* qsrc[2]  = {query, qom};

    char* p = (char*)d_ws;
    _Float16* ApFeat = (_Float16*)ws_carve(p, (size_t)NFC * DIMC * 2);   // packed LN(feat)
    _Float16* bufA   = (_Float16*)ws_carve(p, (size_t)NQC * DIMC * 2);   // packed LN(q) / LN(q_mid)
    _Float16* bufS   = (_Float16*)ws_carve(p, (size_t)NQC * DIMC * 2);   // packed sampled
    _Float16* ApG    = (_Float16*)ws_carve(p, (size_t)NQC * HIDC * 2);   // packed gelu(dwconv)
    _Float16* Bp     = (_Float16*)ws_carve(p, (size_t)DIMC * DIMC * 2);  // packed weights (reused)
    float* vbuf = (float*)ws_carve(p, (size_t)NFC * DIMC * 4);
    float* offb = (float*)ws_carve(p, (size_t)NQC * 48 * 4);
    float* awb  = (float*)ws_carve(p, (size_t)NQC * 24 * 4);
    float* fc1b = (float*)ws_carve(p, (size_t)NQC * HIDC * 4);

    const dim3 gblk(32, 4);   // 4 waves; block covers 128 rows x 64 cols of C

    // Shared: LayerNorm(feat) -> packed A fragments
    ln_pack_kernel<<<NFC, 256, 0, stream>>>(feat, fn_g, fn_b, ApFeat);

    for (int mod = 0; mod < 2; ++mod) {
        float* outm = (float*)d_out + (size_t)mod * NQC * DIMC;

        // value projection: v = LN(feat) @ value_w + b   (4096 x 768)
        pack_w_kernel<<<(DIMC * DIMC) / 256, 256, 0, stream>>>(val_w[mod], Bp, DIMC, DIMC, DIMC);
        gemm_wmma_kernel<<<dim3(NFC / 128, DIMC / 64), gblk, 0, stream>>>(
            ApFeat, Bp, val_b[mod], nullptr, vbuf, DIMC, KB768);

        // LN(query) -> packed
        ln_pack_kernel<<<NQC, 256, 0, stream>>>(qsrc[mod], qn_g[mod], qn_b[mod], bufA);

        // offsets: (21504 x 48), Npad = 64
        pack_w_kernel<<<(DIMC * 64) / 256, 256, 0, stream>>>(off_w[mod], Bp, DIMC, 48, 64);
        gemm_wmma_kernel<<<dim3(NQC / 128, 1), gblk, 0, stream>>>(
            bufA, Bp, off_b[mod], nullptr, offb, 48, KB768);

        // attention weight logits: (21504 x 24), Npad = 64
        pack_w_kernel<<<(DIMC * 64) / 256, 256, 0, stream>>>(aw_w[mod], Bp, DIMC, 24, 64);
        gemm_wmma_kernel<<<dim3(NQC / 128, 1), gblk, 0, stream>>>(
            bufA, Bp, aw_b[mod], nullptr, awb, 24, KB768);

        // softmax + bilinear sampling -> packed A fragments
        msdeform_sample_kernel<<<(NQC * NHC) / 8, 256, 0, stream>>>(vbuf, offb, awb, refp, bufS);

        // out projection + residual -> q_mid (stored in d_out slot)
        pack_w_kernel<<<(DIMC * DIMC) / 256, 256, 0, stream>>>(out_w[mod], Bp, DIMC, DIMC, DIMC);
        gemm_wmma_kernel<<<dim3(NQC / 128, DIMC / 64), gblk, 0, stream>>>(
            bufS, Bp, out_b[mod], qsrc[mod], outm, DIMC, KB768);

        // FFN: LN(q_mid) -> packed
        ln_pack_kernel<<<NQC, 256, 0, stream>>>(outm, ffn_g[mod], ffn_b[mod], bufA);

        // fc1: (21504 x 192)
        pack_w_kernel<<<(DIMC * HIDC) / 256, 256, 0, stream>>>(fc1_w[mod], Bp, DIMC, HIDC, HIDC);
        gemm_wmma_kernel<<<dim3(NQC / 128, HIDC / 64), gblk, 0, stream>>>(
            bufA, Bp, fc1_b[mod], nullptr, fc1b, HIDC, KB768);

        // depthwise conv + GELU -> packed A fragments (K = 192)
        dwconv_gelu_pack_kernel<<<(NQC * 48 + 255) / 256, 256, 0, stream>>>(
            fc1b, dw_k[mod], dw_b[mod], ApG);

        // fc2 + residual(q_mid) -> final output
        pack_w_kernel<<<(HIDC * DIMC) / 256, 256, 0, stream>>>(fc2_w[mod], Bp, HIDC, DIMC, DIMC);
        gemm_wmma_kernel<<<dim3(NQC / 128, DIMC / 64), gblk, 0, stream>>>(
            ApG, Bp, fc2_b[mod], outm, outm, DIMC, KB192);
    }
}
